// LSTMModelWithoutTeacherForce_17712445129194
// MI455X (gfx1250) — compile-verified, matrix operand order checked by reading, not run on hardware
//
#include <hip/hip_runtime.h>
#include <math.h>

// ---- problem constants (from reference) ----
#define VV 32000
#define EE 512
#define HH 512
#define LL 2
#define BB 64
#define TT 256
#define G4H 2048          // 4*H
#define NCHUNK (VV / 16)  // 2000 argmax candidate chunks per row

typedef __attribute__((ext_vector_type(16))) __bf16  v16bf;
typedef __attribute__((ext_vector_type(8)))  float   v8f;
typedef __attribute__((ext_vector_type(4)))  unsigned int u32x4;

union FragBF { v16bf v; u32x4 q[2]; };
union AccF   { v8f v; float f[8]; };

__device__ __forceinline__ unsigned short f32_to_bf16(float x) {
    union { float f; unsigned u; } v; v.f = x;
    unsigned r = v.u + 0x7FFFu + ((v.u >> 16) & 1u);   // round-to-nearest-even
    return (unsigned short)(r >> 16);
}

// Load a 16x32 bf16 WMMA operand fragment.
// row = (m_base | n_base) + (lane & 15); K runs are 16B aligned.
__device__ __forceinline__ v16bf load_frag(const unsigned short* __restrict__ base,
                                           int row, int k0, int lane) {
    FragBF f;
    const unsigned short* p = base + row * 512 + k0 + ((lane >> 4) << 3);
    f.q[0] = *(const u32x4*)(p);
    f.q[1] = *(const u32x4*)(p + 16);
    return f.v;
}

// ---------------- weight conversion / init ----------------
__global__ void cvt_bf16_kernel(const float* __restrict__ src,
                                unsigned short* __restrict__ dst, int n) {
    for (int i = blockIdx.x * blockDim.x + threadIdx.x; i < n;
         i += gridDim.x * blockDim.x)
        dst[i] = f32_to_bf16(src[i]);
}

__global__ void init_state_kernel(float* __restrict__ c,
                                  unsigned short* __restrict__ h, int n) {
    for (int i = blockIdx.x * blockDim.x + threadIdx.x; i < n;
         i += gridDim.x * blockDim.x) { c[i] = 0.0f; h[i] = 0; }
}

__global__ void gather_x0_kernel(const int* __restrict__ x,
                                 const float* __restrict__ emb,
                                 unsigned short* __restrict__ x_bf) {
    int b = blockIdx.x;
    int idx = x[b * TT];                       // x[:,0]
    for (int j = threadIdx.x; j < EE; j += blockDim.x)
        x_bf[b * EE + j] = f32_to_bf16(emb[idx * EE + j]);
}

// ---------------- LSTM gate GEMM: gates = in@Wih^T + h@Whh^T + b ----------------
// grid 64 blocks x 256 threads; each block covers 64 rows x 32 cols of (64 x 2048)
__global__ void gates_kernel(const unsigned short* __restrict__ in_bf,  // 64x512
                             const unsigned short* __restrict__ h_bf,   // 64x512 (prev-step h[l])
                             const unsigned short* __restrict__ wih,    // 2048x512
                             const unsigned short* __restrict__ whh,    // 2048x512
                             const float* __restrict__ b_ih,
                             const float* __restrict__ b_hh,
                             float* __restrict__ gates) {                // 64x2048
    int lane   = threadIdx.x & 31;
    int w      = threadIdx.x >> 5;
    int m_base = (w & 3) * 16;
    int n_base = blockIdx.x * 32 + (w >> 2) * 16;
    int ra = m_base + (lane & 15);
    int rb = n_base + (lane & 15);

    v8f acc = {0.f, 0.f, 0.f, 0.f, 0.f, 0.f, 0.f, 0.f};
    #pragma unroll 4
    for (int k = 0; k < EE; k += 32) {
        v16bf a = load_frag(in_bf, ra, k, lane);
        v16bf b = load_frag(wih,   rb, k, lane);
        acc = __builtin_amdgcn_wmma_f32_16x16x32_bf16(false, a, false, b,
                                                      (short)0, acc, false, false);
    }
    #pragma unroll 4
    for (int k = 0; k < HH; k += 32) {
        v16bf a = load_frag(h_bf, ra, k, lane);
        v16bf b = load_frag(whh,  rb, k, lane);
        acc = __builtin_amdgcn_wmma_f32_16x16x32_bf16(false, a, false, b,
                                                      (short)0, acc, false, false);
    }

    int n = n_base + (lane & 15);
    float bias = b_ih[n] + b_hh[n];
    int rbase = m_base + ((lane >> 4) << 3);
    AccF c; c.v = acc;
    #pragma unroll
    for (int i = 0; i < 8; ++i)
        gates[(rbase + i) * G4H + n] = c.f[i] + bias;
}

// ---------------- LSTM elementwise combine ----------------
__global__ void combine_kernel(const float* __restrict__ gates,  // 64x2048
                               float* __restrict__ c_state,      // 64x512
                               unsigned short* __restrict__ h_bf) { // 64x512
    int t = blockIdx.x * blockDim.x + threadIdx.x;   // 0 .. 64*512-1
    int b = t >> 9, u = t & 511;
    const float* g = gates + b * G4H;
    float ig = 1.0f / (1.0f + expf(-g[u]));
    float fg = 1.0f / (1.0f + expf(-g[u + 512]));
    float gg = tanhf(g[u + 1024]);
    float og = 1.0f / (1.0f + expf(-g[u + 1536]));
    float cn = fg * c_state[t] + ig * gg;
    c_state[t] = cn;
    h_bf[t] = f32_to_bf16(og * tanhf(cn));
}

// ---------------- FC head: logits = h@Wfc^T + b_fc, fused partial argmax ----------------
// grid 1000 blocks x 256 threads; block covers 64 rows x 32 cols of (64 x 32000)
__global__ void fc_kernel(const unsigned short* __restrict__ h_bf,  // 64x512
                          const unsigned short* __restrict__ wfc,   // 32000x512
                          const float* __restrict__ b_fc,
                          float* __restrict__ logits,               // 64x32000
                          float* __restrict__ cval,                 // NCHUNK x 64
                          int* __restrict__ cidx) {
    int lane   = threadIdx.x & 31;
    int w      = threadIdx.x >> 5;
    int m_base = (w & 3) * 16;
    int n_base = blockIdx.x * 32 + (w >> 2) * 16;
    int ra = m_base + (lane & 15);
    int rb = n_base + (lane & 15);

    v8f acc = {0.f, 0.f, 0.f, 0.f, 0.f, 0.f, 0.f, 0.f};
    #pragma unroll 4
    for (int k = 0; k < HH; k += 32) {
        v16bf a = load_frag(h_bf, ra, k, lane);
        v16bf b = load_frag(wfc,  rb, k, lane);
        acc = __builtin_amdgcn_wmma_f32_16x16x32_bf16(false, a, false, b,
                                                      (short)0, acc, false, false);
    }

    int n = n_base + (lane & 15);
    float bias = b_fc[n];
    int rbase = m_base + ((lane >> 4) << 3);
    AccF c; c.v = acc;
    #pragma unroll
    for (int i = 0; i < 8; ++i) {
        float v = c.f[i] + bias;
        c.f[i] = v;
        logits[(rbase + i) * VV + n] = v;
    }

    // per-row argmax over this wave's 16 columns (reduce across 16-lane halves)
    int chunk = blockIdx.x * 2 + (w >> 2);
    #pragma unroll
    for (int i = 0; i < 8; ++i) {
        float v = c.f[i]; int idx = n;
        #pragma unroll
        for (int m = 8; m >= 1; m >>= 1) {
            float ov = __shfl_xor(v, m, 16);
            int   oi = __shfl_xor(idx, m, 16);
            if (ov > v || (ov == v && oi < idx)) { v = ov; idx = oi; }
        }
        if ((lane & 15) == 0) {
            int row = rbase + i;
            cval[chunk * BB + row] = v;
            cidx[chunk * BB + row] = idx;
        }
    }
}

// ---------------- final argmax reduce + next-token embedding gather ----------------
__global__ void argmax_embed_kernel(const float* __restrict__ cval,
                                    const int* __restrict__ cidx,
                                    const float* __restrict__ emb,
                                    unsigned short* __restrict__ x_bf) {
    __shared__ float sv[256];
    __shared__ int   si[256];
    int b = blockIdx.x, t = threadIdx.x;
    float best = -INFINITY; int bidx = 0x7FFFFFFF;
    for (int c = t; c < NCHUNK; c += 256) {
        float v = cval[c * BB + b];
        int   i = cidx[c * BB + b];
        if (v > best || (v == best && i < bidx)) { best = v; bidx = i; }
    }
    sv[t] = best; si[t] = bidx;
    __syncthreads();
    for (int s = 128; s > 0; s >>= 1) {
        if (t < s) {
            float ov = sv[t + s]; int oi = si[t + s];
            if (ov > sv[t] || (ov == sv[t] && oi < si[t])) { sv[t] = ov; si[t] = oi; }
        }
        __syncthreads();
    }
    int idx = si[0];
    for (int j = t; j < EE; j += 256)
        x_bf[b * EE + j] = f32_to_bf16(emb[idx * EE + j]);
}

// ---------------- host driver ----------------
extern "C" void kernel_launch(void* const* d_in, const int* in_sizes, int n_in,
                              void* d_out, int out_size, void* d_ws, size_t ws_size,
                              hipStream_t stream) {
    (void)in_sizes; (void)n_in; (void)out_size; (void)ws_size;
    const int*   x    = (const int*)d_in[0];
    const float* emb  = (const float*)d_in[1];
    const float* Wih  = (const float*)d_in[2];   // L x 4H x E
    const float* Whh  = (const float*)d_in[3];   // L x 4H x H
    const float* bih  = (const float*)d_in[4];   // L x 4H
    const float* bhh  = (const float*)d_in[5];   // L x 4H
    const float* Wfc  = (const float*)d_in[6];   // V x H
    const float* bfc  = (const float*)d_in[7];   // V
    float* logits = (float*)d_out;               // B x V

    // workspace carve-up (256B aligned)
    char* p = (char*)d_ws;
    auto carve = [&](size_t bytes) { char* r = p; p += (bytes + 255) & ~(size_t)255; return r; };
    unsigned short* Wih_bf = (unsigned short*)carve((size_t)LL * G4H * EE * 2);
    unsigned short* Whh_bf = (unsigned short*)carve((size_t)LL * G4H * HH * 2);
    unsigned short* Wfc_bf = (unsigned short*)carve((size_t)VV * HH * 2);
    unsigned short* h_bf   = (unsigned short*)carve((size_t)LL * BB * HH * 2);
    float*          c_st   = (float*)carve((size_t)LL * BB * HH * 4);
    unsigned short* x_bf   = (unsigned short*)carve((size_t)BB * EE * 2);
    float*          gates  = (float*)carve((size_t)BB * G4H * 4);
    float*          cval   = (float*)carve((size_t)NCHUNK * BB * 4);
    int*            cidx   = (int*)carve((size_t)NCHUNK * BB * 4);

    // one-time (per call) weight conversion + state init
    cvt_bf16_kernel<<<1024, 256, 0, stream>>>(Wih, Wih_bf, LL * G4H * EE);
    cvt_bf16_kernel<<<1024, 256, 0, stream>>>(Whh, Whh_bf, LL * G4H * HH);
    cvt_bf16_kernel<<<4096, 256, 0, stream>>>(Wfc, Wfc_bf, VV * HH);
    init_state_kernel<<<256, 256, 0, stream>>>(c_st, h_bf, LL * BB * HH);
    gather_x0_kernel<<<BB, 256, 0, stream>>>(x, emb, x_bf);

    for (int t = 0; t < TT; ++t) {
        for (int l = 0; l < LL; ++l) {
            const unsigned short* in_bf = (l == 0) ? x_bf : (h_bf + (size_t)(l - 1) * BB * HH);
            unsigned short* hl = h_bf + (size_t)l * BB * HH;
            float* cl = c_st + (size_t)l * BB * HH;
            gates_kernel<<<G4H / 32, 256, 0, stream>>>(
                in_bf, hl,
                Wih_bf + (size_t)l * G4H * EE, Whh_bf + (size_t)l * G4H * HH,
                bih + (size_t)l * G4H, bhh + (size_t)l * G4H, gates);
            combine_kernel<<<(BB * HH) / 256, 256, 0, stream>>>(gates, cl, hl);
        }
        fc_kernel<<<VV / 32, 256, 0, stream>>>(
            h_bf + (size_t)(LL - 1) * BB * HH, Wfc_bf, bfc, logits, cval, cidx);
        argmax_embed_kernel<<<BB, 256, 0, stream>>>(cval, cidx, emb, x_bf);
    }
}
